// PCM_c3_77567109366260
// MI455X (gfx1250) — matrix operand order repaired and checked by reading, not compiled, not run on hardware
//
#include <hip/hip_runtime.h>
#include <hip/hip_bf16.h>

// ---------------- problem constants ----------------
constexpr int T  = 400;
constexpr int B  = 512;
constexpr int XD = 88;
constexpr int HD = 256;
constexpr int ZD = 64;

constexpr int KPE = 352;  // prior/enc K  (344 -> pad 352)
constexpr int KD  = 672;  // dec K        (664 -> pad 672)
constexpr int KR  = 160;  // rnn K        (152 -> pad 160)
constexpr int XDP = 96;   // dm/ds N pad  (88  -> pad 96)

typedef __attribute__((ext_vector_type(16))) __bf16 v16bf;
typedef __attribute__((ext_vector_type(8)))  __bf16 v8bf;
typedef __attribute__((ext_vector_type(2)))  __bf16 bf16x2;
typedef __attribute__((ext_vector_type(8)))  float  v8f;

// ---------------- WMMA tile helper ----------------
// C/D: VGPR r -> row = r + 8*(lane>>4), col = n0 + (lane&15)
// A 16x32 bf16 fragment per lane = two contiguous 8-elem runs:
//   A[m][k0 + 8*half .. +8)  and  A[m][k0 + 16 + 8*half .. +8)   (2x ds_load_b128)
// B fragment: pre-shuffled weights, lane's 16 elems contiguous (2x global_load_b128)
//   layout: [kb][ntile][lane][16] with elem j <-> k = kb*32 + 8*(lane>>4) + (j&7) + 16*(j>>3),
//           n = ntile*16 + (lane&15)
__device__ __forceinline__ v8f wmma_tile(const __bf16* A, int lda,
                                         const __bf16* Bshuf, int ntiles, int ntile,
                                         int K, int lane, v8f acc) {
  const int m    = lane & 15;
  const int half = lane >> 4;
  const __bf16* arow  = A + m * lda + half * 8;
  const __bf16* bbase = Bshuf + ((size_t)ntile * 32 + lane) * 16;
  const int kb_stride = ntiles * 32 * 16;
  for (int k0 = 0; k0 < K; k0 += 32) {
    v8bf a_lo = *(const v8bf*)(arow + k0);
    v8bf a_hi = *(const v8bf*)(arow + k0 + 16);
    v16bf av = __builtin_shufflevector(a_lo, a_hi,
                 0,1,2,3,4,5,6,7,8,9,10,11,12,13,14,15);
    v16bf bv = *(const v16bf*)bbase;
    bbase += kb_stride;
    acc = __builtin_amdgcn_wmma_f32_16x16x32_bf16(false, av, false, bv,
                                                  (short)0, acc, false, false);
  }
  return acc;
}

__device__ __forceinline__ float softplus_f(float x) {
  return x > 20.f ? x : log1pf(__expf(x));
}

// ---------------- shared memory layout (~60 KB) ----------------
struct SMem {
  __bf16 A1[16][KPE];           // [xp|h] then [x|h]
  __bf16 A2[16][KD];            // [xp|z|h|h1]  (h at col 152, 256 wide)
  union { __bf16 Act[16][HD];   // pr / en / de as bf16 A-matrix
          __bf16 Ar[16][KR]; }; // [x|z] for RNN (after de consumed)
  union {
    struct { float Pm[16][ZD]; float Ps[16][ZD];
             float Em[16][ZD]; float Es[16][ZD]; } kl;
    struct { float Dm[16][XDP]; float Ds[16][XDP]; } dd;
  } u;
  float Z[16][ZD];
  float red;
};

// ---------------- per-timestep kernel ----------------
__global__ __launch_bounds__(256)
void vrnn_step(const float* __restrict__ x_t, const float* __restrict__ xp_t,
               const float* __restrict__ eps_t,
               const __bf16* Wp,  const __bf16* We,  const __bf16* Wd,
               const __bf16* Wih, const __bf16* Whh,
               const __bf16* Wpm, const __bf16* Wps,
               const __bf16* Wem, const __bf16* Wes,
               const __bf16* Wdm, const __bf16* Wds,
               const float* b_prior, const float* b_pm, const float* b_ps,
               const float* b_enc,   const float* b_em, const float* b_es,
               const float* b_dec,   const float* b_dm, const float* b_ds,
               const float* __restrict__ h, const float* __restrict__ h1,
               float* __restrict__ hnew,
               float* kld_acc, float* nll_acc, float* dm_last,
               float* ems_t, float* ess_t, float* dms_t, float* dss_t,
               int is_last) {
  __shared__ SMem sm;
  const int tid  = threadIdx.x;
  const int wave = tid >> 5;
  const int lane = tid & 31;
  const int half = lane >> 4;
  const int nn   = lane & 15;
  const int m0   = blockIdx.x * 16;

  // ---- Phase 0: stage A1=[xp|h] and A2=[xp|z(0)|h|h1] as bf16 (paired cols) ----
  for (int i = tid; i < 16 * (KPE / 2); i += 256) {
    int r = i / (KPE / 2), c = (i % (KPE / 2)) * 2;
    float v0 = 0.f, v1 = 0.f;
    if (c < XD) {
      if (xp_t) { v0 = xp_t[(m0 + r) * XD + c]; v1 = xp_t[(m0 + r) * XD + c + 1]; }
    } else if (c < XD + HD) {
      v0 = h[(m0 + r) * HD + (c - XD)];
      v1 = h[(m0 + r) * HD + (c - XD) + 1];
    }
    bf16x2 p = { (__bf16)v0, (__bf16)v1 };
    *(bf16x2*)&sm.A1[r][c] = p;
  }
  for (int i = tid; i < 16 * (KD / 2); i += 256) {
    int r = i / (KD / 2), c = (i % (KD / 2)) * 2;
    float v0 = 0.f, v1 = 0.f;
    if (c < 88) {
      if (xp_t) { v0 = xp_t[(m0 + r) * XD + c]; v1 = xp_t[(m0 + r) * XD + c + 1]; }
    } else if (c < 152) {
      // z, filled later
    } else if (c < 408) {
      v0 = h[(m0 + r) * HD + (c - 152)];
      v1 = h[(m0 + r) * HD + (c - 152) + 1];
    } else if (c < 664) {
      v0 = h1[(m0 + r) * HD + (c - 408)];
      v1 = h1[(m0 + r) * HD + (c - 408) + 1];
    }
    bf16x2 p = { (__bf16)v0, (__bf16)v1 };
    *(bf16x2*)&sm.A2[r][c] = p;
  }
  __syncthreads();

  // ---- Phase 1: pr = relu([xp|h] @ W_prior + b) -> Act ----
  for (int j = 0; j < 2; ++j) {
    int nt = wave * 2 + j;
    v8f acc = {};
    acc = wmma_tile(&sm.A1[0][0], KPE, Wp, HD / 16, nt, KPE, lane, acc);
#pragma unroll
    for (int r = 0; r < 8; ++r) {
      int row = r + 8 * half, n = nt * 16 + nn;
      float v = acc[r] + b_prior[n];
      sm.Act[row][n] = (__bf16)(v > 0.f ? v : 0.f);
    }
  }
  __syncthreads();

  // ---- Phase 2: pm / softplus(ps) from pr ----
  {
    int nt = wave & 3;
    v8f acc = {};
    acc = wmma_tile(&sm.Act[0][0], HD, (wave < 4 ? Wpm : Wps), ZD / 16, nt, HD, lane, acc);
#pragma unroll
    for (int r = 0; r < 8; ++r) {
      int row = r + 8 * half, n = nt * 16 + nn;
      if (wave < 4) sm.u.kl.Pm[row][n] = acc[r] + b_pm[n];
      else          sm.u.kl.Ps[row][n] = softplus_f(acc[r] + b_ps[n]);
    }
  }
  __syncthreads();

  // ---- Phase 3: A1 cols 0..87 := x_t ; en = relu([x|h] @ W_enc + b) -> Act ----
  for (int i = tid; i < 16 * (XD / 2); i += 256) {
    int r = i / (XD / 2), c = (i % (XD / 2)) * 2;
    bf16x2 p = { (__bf16)x_t[(m0 + r) * XD + c], (__bf16)x_t[(m0 + r) * XD + c + 1] };
    *(bf16x2*)&sm.A1[r][c] = p;
  }
  __syncthreads();
  for (int j = 0; j < 2; ++j) {
    int nt = wave * 2 + j;
    v8f acc = {};
    acc = wmma_tile(&sm.A1[0][0], KPE, We, HD / 16, nt, KPE, lane, acc);
#pragma unroll
    for (int r = 0; r < 8; ++r) {
      int row = r + 8 * half, n = nt * 16 + nn;
      float v = acc[r] + b_enc[n];
      sm.Act[row][n] = (__bf16)(v > 0.f ? v : 0.f);
    }
  }
  __syncthreads();

  // ---- Phase 4: em / softplus(es) from en; store to LDS + global ----
  {
    int nt = wave & 3;
    v8f acc = {};
    acc = wmma_tile(&sm.Act[0][0], HD, (wave < 4 ? Wem : Wes), ZD / 16, nt, HD, lane, acc);
#pragma unroll
    for (int r = 0; r < 8; ++r) {
      int row = r + 8 * half, n = nt * 16 + nn;
      if (wave < 4) {
        float v = acc[r] + b_em[n];
        sm.u.kl.Em[row][n] = v;
        ems_t[(m0 + row) * ZD + n] = v;
      } else {
        float v = softplus_f(acc[r] + b_es[n]);
        sm.u.kl.Es[row][n] = v;
        ess_t[(m0 + row) * ZD + n] = v;
      }
    }
  }
  __syncthreads();

  // ---- Phase 5: z = eps*es + em ; KLD reduction ----
  if (tid == 0) sm.red = 0.f;
  __syncthreads();
  {
    float kpart = 0.f;
    for (int i = tid; i < 16 * ZD; i += 256) {
      int r = i / ZD, c = i % ZD;
      float em = sm.u.kl.Em[r][c], es = sm.u.kl.Es[r][c];
      float pm = sm.u.kl.Pm[r][c], ps = sm.u.kl.Ps[r][c];
      float z  = eps_t[(m0 + r) * ZD + c] * es + em;
      sm.Z[r][c]       = z;
      sm.A2[r][XD + c] = (__bf16)z;
      float d = em - pm;
      kpart += __logf(ps) - __logf(es) + (es * es + d * d) / (2.f * ps * ps) - 0.5f;
    }
    atomicAdd(&sm.red, kpart);
  }
  __syncthreads();
  if (tid == 0) atomicAdd(kld_acc, sm.red);
  __syncthreads();

  // ---- Phase 6: de = relu([xp|z|h|h1] @ W_dec + b) -> Act ----
  for (int j = 0; j < 2; ++j) {
    int nt = wave * 2 + j;
    v8f acc = {};
    acc = wmma_tile(&sm.A2[0][0], KD, Wd, HD / 16, nt, KD, lane, acc);
#pragma unroll
    for (int r = 0; r < 8; ++r) {
      int row = r + 8 * half, n = nt * 16 + nn;
      float v = acc[r] + b_dec[n];
      sm.Act[row][n] = (__bf16)(v > 0.f ? v : 0.f);
    }
  }
  __syncthreads();

  // ---- Phase 7: dm / ds logits (12 n-tile jobs over 8 waves) ----
  for (int j = wave; j < 12; j += 8) {
    bool isDm = (j < 6);
    int nt = isDm ? j : j - 6;
    v8f acc = {};
    acc = wmma_tile(&sm.Act[0][0], HD, (isDm ? Wdm : Wds), XDP / 16, nt, HD, lane, acc);
#pragma unroll
    for (int r = 0; r < 8; ++r) {
      int row = r + 8 * half, n = nt * 16 + nn;
      float bias = (n < XD) ? (isDm ? b_dm[n] : b_ds[n]) : 0.f;
      if (isDm) sm.u.dd.Dm[row][n] = acc[r] + bias;
      else      sm.u.dd.Ds[row][n] = acc[r] + bias;
    }
  }
  __syncthreads();

  // ---- Phase 8: sigmoid + NLL + dms; row softmax -> dss ----
  if (tid == 0) sm.red = 0.f;
  __syncthreads();
  {
    float npart = 0.f;
    for (int i = tid; i < 16 * XD; i += 256) {
      int r = i / XD, c = i % XD;
      float dmv = 1.f / (1.f + __expf(-sm.u.dd.Dm[r][c]));
      dms_t[(m0 + r) * XD + c] = dmv;
      if (is_last) dm_last[(m0 + r) * XD + c] = dmv;
      float xv = x_t[(m0 + r) * XD + c];
      npart -= xv * __logf(dmv) + (1.f - xv) * log1pf(-dmv);
    }
    atomicAdd(&sm.red, npart);
  }
  if (tid < 16) {
    float mx = -1e30f;
    for (int c = 0; c < XD; ++c) mx = fmaxf(mx, sm.u.dd.Ds[tid][c]);
    float s = 0.f;
    for (int c = 0; c < XD; ++c) s += __expf(sm.u.dd.Ds[tid][c] - mx);
    float inv = 1.f / s;
    for (int c = 0; c < XD; ++c)
      dss_t[(m0 + tid) * XD + c] = __expf(sm.u.dd.Ds[tid][c] - mx) * inv;
  }
  __syncthreads();
  if (tid == 0) atomicAdd(nll_acc, sm.red);
  __syncthreads();

  // ---- Phase 9: h_new = tanh([x|z] @ W_ih + h @ W_hh) ----
  for (int i = tid; i < 16 * (KR / 2); i += 256) {  // Ar aliases Act (de consumed)
    int r = i / (KR / 2), c = (i % (KR / 2)) * 2;
    float v0 = 0.f, v1 = 0.f;
    if (c < XD) {
      v0 = x_t[(m0 + r) * XD + c];
      v1 = x_t[(m0 + r) * XD + c + 1];
    } else if (c < XD + ZD) {
      v0 = sm.Z[r][c - XD];
      v1 = sm.Z[r][c - XD + 1];
    }
    bf16x2 p = { (__bf16)v0, (__bf16)v1 };
    *(bf16x2*)&sm.Ar[r][c] = p;
  }
  __syncthreads();
  for (int j = 0; j < 2; ++j) {
    int nt = wave * 2 + j;
    v8f acc = {};
    acc = wmma_tile(&sm.Ar[0][0], KR, Wih, HD / 16, nt, KR, lane, acc);
    acc = wmma_tile(&sm.A2[0][152], KD, Whh, HD / 16, nt, HD, lane, acc);  // h region of A2
#pragma unroll
    for (int r = 0; r < 8; ++r) {
      int row = r + 8 * half, n = nt * 16 + nn;
      hnew[(m0 + row) * HD + n] = tanhf(acc[r]);
    }
  }
}

// ---------------- prep kernels ----------------
__global__ void init_kernel(float* hbuf, float* out01) {
  int i = blockIdx.x * blockDim.x + threadIdx.x;
  if (i < 3 * B * HD) hbuf[i] = 0.f;
  if (i < 2) out01[i] = 0.f;
}

// fp32 [K][N] -> bf16 fragment-shuffled [Kp/32][Np/16][32][16], zero-padded.
//   elem j of lane <-> k = kb*32 + 8*(lane>>4) + (j&7) + 16*(j>>3),  n = nt*16 + (lane&15)
__global__ void cvt_shuf_kernel(const float* __restrict__ src, __bf16* __restrict__ dst,
                                int K, int N, int Kp, int Np) {
  int i = blockIdx.x * blockDim.x + threadIdx.x;
  if (i >= Kp * Np) return;
  int j    = i & 15;
  int lane = (i >> 4) & 31;
  int nt   = (i >> 9) % (Np / 16);
  int kb   = (i >> 9) / (Np / 16);
  int k = kb * 32 + (lane >> 4) * 8 + (j & 7) + (j >> 3) * 16;
  int n = nt * 16 + (lane & 15);
  float v = (k < K && n < N) ? src[k * N + n] : 0.f;
  dst[i] = (__bf16)v;
}

// ---------------- host launch ----------------
extern "C" void kernel_launch(void* const* d_in, const int* in_sizes, int n_in,
                              void* d_out, int out_size, void* d_ws, size_t ws_size,
                              hipStream_t stream) {
  const float* x       = (const float*)d_in[0];
  const float* eps     = (const float*)d_in[1];
  const float* W_prior = (const float*)d_in[2];  const float* b_prior = (const float*)d_in[3];
  const float* W_pm    = (const float*)d_in[4];  const float* b_pm    = (const float*)d_in[5];
  const float* W_ps    = (const float*)d_in[6];  const float* b_ps    = (const float*)d_in[7];
  const float* W_enc   = (const float*)d_in[8];  const float* b_enc   = (const float*)d_in[9];
  const float* W_em    = (const float*)d_in[10]; const float* b_em    = (const float*)d_in[11];
  const float* W_es    = (const float*)d_in[12]; const float* b_es    = (const float*)d_in[13];
  const float* W_dec   = (const float*)d_in[14]; const float* b_dec   = (const float*)d_in[15];
  const float* W_dm    = (const float*)d_in[16]; const float* b_dm    = (const float*)d_in[17];
  const float* W_ds    = (const float*)d_in[18]; const float* b_ds    = (const float*)d_in[19];
  const float* W_ih    = (const float*)d_in[20];
  const float* W_hh    = (const float*)d_in[21];

  float* out = (float*)d_out;

  // workspace: 3 h-state buffers + bf16 fragment-shuffled weights
  float* hbuf = (float*)d_ws;
  float* hS[3] = { hbuf, hbuf + (size_t)B * HD, hbuf + (size_t)2 * B * HD };
  __bf16* wb = (__bf16*)(hbuf + (size_t)3 * B * HD);
  __bf16* Wp  = wb; wb += KPE * HD;
  __bf16* We  = wb; wb += KPE * HD;
  __bf16* Wd  = wb; wb += KD  * HD;
  __bf16* Wih = wb; wb += KR  * HD;
  __bf16* Whh = wb; wb += HD  * HD;
  __bf16* Wpm = wb; wb += HD * ZD;
  __bf16* Wps = wb; wb += HD * ZD;
  __bf16* Wem = wb; wb += HD * ZD;
  __bf16* Wes = wb; wb += HD * ZD;
  __bf16* Wdm = wb; wb += HD * XDP;
  __bf16* Wds = wb; wb += HD * XDP;

  // prep: zero loss accumulators + h state; convert+shuffle weights to bf16
  init_kernel<<<(3 * B * HD + 255) / 256, 256, 0, stream>>>(hbuf, out);
  auto cvt = [&](const float* s, __bf16* d, int K, int N, int Kp, int Np) {
    cvt_shuf_kernel<<<(Kp * Np + 255) / 256, 256, 0, stream>>>(s, d, K, N, Kp, Np);
  };
  cvt(W_prior, Wp,  344, HD, KPE, HD);
  cvt(W_enc,   We,  344, HD, KPE, HD);
  cvt(W_dec,   Wd,  664, HD, KD,  HD);
  cvt(W_ih,    Wih, 152, HD, KR,  HD);
  cvt(W_hh,    Whh, HD,  HD, HD,  HD);
  cvt(W_pm,    Wpm, HD,  ZD, HD,  ZD);
  cvt(W_ps,    Wps, HD,  ZD, HD,  ZD);
  cvt(W_em,    Wem, HD,  ZD, HD,  ZD);
  cvt(W_es,    Wes, HD,  ZD, HD,  ZD);
  cvt(W_dm,    Wdm, HD,  XD, HD,  XDP);
  cvt(W_ds,    Wds, HD,  XD, HD,  XDP);

  // output layout: kld, nll, dm_last, ems, ess, dms, dss
  float* kld_acc = out;
  float* nll_acc = out + 1;
  float* dm_last = out + 2;
  float* ems = dm_last + (size_t)B * XD;
  float* ess = ems + (size_t)T * B * ZD;
  float* dms = ess + (size_t)T * B * ZD;
  float* dss = dms + (size_t)T * B * XD;

  int ih = 0, ih1 = 1, io = 2;  // (h, h1, h_new) buffer roles
  for (int t = 0; t < T; ++t) {
    const float* x_t  = x + (size_t)t * B * XD;
    const float* xp_t = (t == 0) ? nullptr : x + (size_t)(t - 1) * B * XD;
    const float* e_t  = eps + (size_t)t * B * ZD;
    vrnn_step<<<B / 16, 256, 0, stream>>>(
        x_t, xp_t, e_t,
        Wp, We, Wd, Wih, Whh, Wpm, Wps, Wem, Wes, Wdm, Wds,
        b_prior, b_pm, b_ps, b_enc, b_em, b_es, b_dec, b_dm, b_ds,
        hS[ih], hS[ih1], hS[io],
        kld_acc, nll_acc, dm_last,
        ems + (size_t)t * B * ZD, ess + (size_t)t * B * ZD,
        dms + (size_t)t * B * XD, dss + (size_t)t * B * XD,
        (t == T - 1) ? 1 : 0);
    int tmp = ih1; ih1 = ih; ih = io; io = tmp;  // (h,h1,out) <- (h_new, h, h1)
  }
}
